// Epipolar_48713519071822
// MI455X (gfx1250) — matrix operand order repaired and checked by reading, not compiled, not run on hardware
//
#include <hip/hip_runtime.h>
#include <math.h>

typedef __attribute__((ext_vector_type(16))) _Float16 v16h;
typedef __attribute__((ext_vector_type(8)))  _Float16 v8h;
typedef __attribute__((ext_vector_type(8)))  float    v8f;

#define HH 64
#define WW 64
#define CC 256
#define SS 32
#define EPSL 0.001f
#define SPAD 264   // LDS row stride in halves: 264*2B=528B=132 banks -> 4-bank skew per row

// ---------------------------------------------------------------------------
// Kernel 1: fundamental matrix F[n] = skew(P2 @ Cc) @ (P2 @ pinv(P1))
//   pinv(P1) = P1^T (P1 P1^T)^-1  (full row rank)
//   Cc = unit null vector of P1 via 3x3 cofactors  (== SVD Vt[-1] up to sign)
// ---------------------------------------------------------------------------
__global__ void epi_compute_F(const float* __restrict__ P1,
                              const float* __restrict__ P2,
                              float* __restrict__ Fout, int Nb) {
    int n = blockIdx.x * blockDim.x + threadIdx.x;
    if (n >= Nb) return;
    const float* A = P1 + n * 12;   // 3x4 row-major
    const float* B = P2 + n * 12;

    // M = A A^T (3x3)
    float M[3][3];
    for (int i = 0; i < 3; ++i)
        for (int j = 0; j < 3; ++j) {
            float s = 0.f;
            for (int k = 0; k < 4; ++k) s += A[i*4+k] * A[j*4+k];
            M[i][j] = s;
        }
    // inverse of M
    float d00 = M[1][1]*M[2][2] - M[1][2]*M[2][1];
    float d01 = M[1][0]*M[2][2] - M[1][2]*M[2][0];
    float d02 = M[1][0]*M[2][1] - M[1][1]*M[2][0];
    float det = M[0][0]*d00 - M[0][1]*d01 + M[0][2]*d02;
    float id = 1.0f / det;
    float Mi[3][3];
    Mi[0][0] =  d00*id; Mi[0][1] = -(M[0][1]*M[2][2]-M[0][2]*M[2][1])*id; Mi[0][2] =  (M[0][1]*M[1][2]-M[0][2]*M[1][1])*id;
    Mi[1][0] = -d01*id; Mi[1][1] =  (M[0][0]*M[2][2]-M[0][2]*M[2][0])*id; Mi[1][2] = -(M[0][0]*M[1][2]-M[0][2]*M[1][0])*id;
    Mi[2][0] =  d02*id; Mi[2][1] = -(M[0][0]*M[2][1]-M[0][1]*M[2][0])*id; Mi[2][2] =  (M[0][0]*M[1][1]-M[0][1]*M[1][0])*id;

    // pinv(P1)[k][j] = sum_i A[i][k] * Mi[i][j]   (4x3)
    float Pi[4][3];
    for (int k = 0; k < 4; ++k)
        for (int j = 0; j < 3; ++j) {
            float s = 0.f;
            for (int i = 0; i < 3; ++i) s += A[i*4+k] * Mi[i][j];
            Pi[k][j] = s;
        }

    // null vector of A via cofactors
    float v[4];
    {
        auto det3 = [&](int i, int j, int k) {
            float a0=A[0*4+i], a1=A[0*4+j], a2=A[0*4+k];
            float b0=A[1*4+i], b1=A[1*4+j], b2=A[1*4+k];
            float c0=A[2*4+i], c1=A[2*4+j], c2=A[2*4+k];
            return a0*(b1*c2-b2*c1) - a1*(b0*c2-b2*c0) + a2*(b0*c1-b1*c0);
        };
        v[0] =  det3(1,2,3);
        v[1] = -det3(0,2,3);
        v[2] =  det3(0,1,3);
        v[3] = -det3(0,1,2);
        float nrm = sqrtf(v[0]*v[0]+v[1]*v[1]+v[2]*v[2]+v[3]*v[3]);
        float inv = 1.0f / nrm;
        for (int i = 0; i < 4; ++i) v[i] *= inv;
    }

    // e2 = B @ v
    float e2[3];
    for (int i = 0; i < 3; ++i) {
        float s = 0.f;
        for (int k = 0; k < 4; ++k) s += B[i*4+k] * v[k];
        e2[i] = s;
    }
    // T = B @ Pi (3x3)
    float T[3][3];
    for (int i = 0; i < 3; ++i)
        for (int j = 0; j < 3; ++j) {
            float s = 0.f;
            for (int k = 0; k < 4; ++k) s += B[i*4+k] * Pi[k][j];
            T[i][j] = s;
        }
    // F = skew(e2) @ T
    float* F = Fout + n * 9;
    F[0] = -e2[2]*T[1][0] + e2[1]*T[2][0];
    F[1] = -e2[2]*T[1][1] + e2[1]*T[2][1];
    F[2] = -e2[2]*T[1][2] + e2[1]*T[2][2];
    F[3] =  e2[2]*T[0][0] - e2[0]*T[2][0];
    F[4] =  e2[2]*T[0][1] - e2[0]*T[2][1];
    F[5] =  e2[2]*T[0][2] - e2[0]*T[2][2];
    F[6] = -e2[1]*T[0][0] + e2[0]*T[1][0];
    F[7] = -e2[1]*T[0][1] + e2[0]*T[1][1];
    F[8] = -e2[1]*T[0][2] + e2[0]*T[1][2];
}

// ---------------------------------------------------------------------------
// Kernel 2: NCHW f32 -> NHWC f16 repack (write-coalesced)
// ---------------------------------------------------------------------------
__global__ void epi_nchw_to_nhwc_f16(const float* __restrict__ in,
                                     _Float16* __restrict__ out, int total) {
    int idx = blockIdx.x * blockDim.x + threadIdx.x;
    if (idx >= total) return;
    int c = idx & (CC - 1);
    int w = (idx >> 8) & (WW - 1);
    int h = (idx >> 14) & (HH - 1);
    int n = idx >> 20;
    out[idx] = (_Float16)in[(((n * CC + c) * HH + h) * WW) + w];
}

// ---------------------------------------------------------------------------
// Kernel 3: fused epipolar attention. One wave (32 lanes) per pixel.
//   lane s: bilinear-gathers sampled[s][0..255] (f16) into LDS
//   phase 1: scores = sampled(32x256) @ q(256) via 16x v_wmma_f32_16x16x32_f16
//   softmax over S=32 with a single shfl_xor(16) exchange
//   phase 2: out[c] = sum_s w[s]*sampled[s][c], lane owns 8 channels
// ---------------------------------------------------------------------------
__global__ __launch_bounds__(32)
void epi_attn(const _Float16* __restrict__ f1h,   // NHWC f16
              const _Float16* __restrict__ f2h,   // NHWC f16
              const float* __restrict__ Fm,       // [Nb,9]
              float* __restrict__ out) {          // NCHW f32
    const int pix  = blockIdx.x;
    const int lane = threadIdx.x;
    const int n  = pix >> 12;       // H*W = 4096
    const int hw = pix & 4095;
    const int h  = hw >> 6;
    const int w  = hw & 63;

    __shared__ _Float16 smem[SS * SPAD];
    __shared__ float    wgt[SS];

    // ---- epipolar line & sample location (uniform across lanes, per-lane t)
    const float* Fp = Fm + n * 9;
    const float gx = 4.0f * (float)w + 1.5f;
    const float gy = 4.0f * (float)h + 1.5f;
    const float la = Fp[0]*gx + Fp[1]*gy + Fp[2];
    const float lb = Fp[3]*gx + Fp[4]*gy + Fp[5];
    const float lc = Fp[6]*gx + Fp[7]*gy + Fp[8];
    const float xmin = 1.5f, xmax = 253.5f, ymin = 1.5f, ymax = 253.5f, tol = 0.01f;
    const float a_s = (fabsf(la) < EPSL) ? EPSL : la;
    const float b_s = (fabsf(lb) < EPSL) ? EPSL : lb;
    float cxs[4] = { xmin, xmax, -(lc + lb*ymin)/a_s, -(lc + lb*ymax)/a_s };
    float cys[4] = { -(lc + la*xmin)/b_s, -(lc + la*xmax)/b_s, ymin, ymax };
    bool  val[4];
    float rx[4], ry[4];
    #pragma unroll
    for (int i = 0; i < 4; ++i) {
        val[i] = (cxs[i] >= xmin - tol) && (cxs[i] <= xmax + tol) &&
                 (cys[i] >= ymin - tol) && (cys[i] <= ymax + tol);
        rx[i] = val[i] ? cxs[i] : (xmin - 10000.0f);
        ry[i] = val[i] ? cys[i] : (ymin - 10000.0f);
    }
    int p0 = -1, p1 = -1;                       // stable argsort(!valid)[:2]
    #pragma unroll
    for (int i = 0; i < 4; ++i) if (val[i])  { if (p0 < 0) p0 = i; else if (p1 < 0) p1 = i; }
    #pragma unroll
    for (int i = 0; i < 4; ++i) if (!val[i]) { if (p0 < 0) p0 = i; else if (p1 < 0) p1 = i; }
    const float inv_rng = 2.0f / (xmax - xmin);
    const float sx = (rx[p0] - xmin) * inv_rng - 1.0f;
    const float sy = (ry[p0] - ymin) * inv_rng - 1.0f;
    const float ex = (rx[p1] - xmin) * inv_rng - 1.0f;
    const float ey = (ry[p1] - ymin) * inv_rng - 1.0f;

    const float t  = (float)lane * (1.0f / 31.0f);
    const float fx = sx + t * (ex - sx);
    const float fy = sy + t * (ey - sy);
    const float x  = (fx + 1.0f) * 0.5f * (float)(WW - 1);
    const float y  = (fy + 1.0f) * 0.5f * (float)(HH - 1);
    const float x0f = floorf(x), y0f = floorf(y);
    const float wx = x - x0f, wy = y - y0f;
    const int x0 = (int)x0f, y0 = (int)y0f, x1 = x0 + 1, y1 = y0 + 1;
    const bool bx0 = (x0 >= 0) & (x0 <= WW-1), bx1 = (x1 >= 0) & (x1 <= WW-1);
    const bool by0 = (y0 >= 0) & (y0 <= HH-1), by1 = (y1 >= 0) & (y1 <= HH-1);
    const int x0c = min(max(x0, 0), WW-1), x1c = min(max(x1, 0), WW-1);
    const int y0c = min(max(y0, 0), HH-1), y1c = min(max(y1, 0), HH-1);
    const float w00 = (1.0f-wy)*(1.0f-wx) * ((by0 & bx0) ? 1.0f : 0.0f);
    const float w01 = (1.0f-wy)*wx        * ((by0 & bx1) ? 1.0f : 0.0f);
    const float w10 = wy*(1.0f-wx)        * ((by1 & bx0) ? 1.0f : 0.0f);
    const float w11 = wy*wx               * ((by1 & bx1) ? 1.0f : 0.0f);

    // ---- gather + bilinear blend into LDS (f16), lane owns row s=lane
    const _Float16* base = f2h + ((size_t)n << 20);          // n*H*W*C
    const v8h* p00 = (const v8h*)(base + ((((size_t)y0c << 6) + x0c) << 8));
    const v8h* p01 = (const v8h*)(base + ((((size_t)y0c << 6) + x1c) << 8));
    const v8h* p10 = (const v8h*)(base + ((((size_t)y1c << 6) + x0c) << 8));
    const v8h* p11 = (const v8h*)(base + ((((size_t)y1c << 6) + x1c) << 8));
    _Float16* srow = &smem[lane * SPAD];
    #pragma unroll 4
    for (int cb = 0; cb < CC / 8; ++cb) {
        v8h t00 = p00[cb], t01 = p01[cb], t10 = p10[cb], t11 = p11[cb];
        v8h o;
        #pragma unroll
        for (int j = 0; j < 8; ++j) {
            float v = w00 * (float)t00[j] + w01 * (float)t01[j]
                    + w10 * (float)t10[j] + w11 * (float)t11[j];
            o[j] = (_Float16)v;
        }
        *(v8h*)(srow + cb * 8) = o;
    }
    __syncthreads();

    // ---- phase 1: scores via WMMA (q replicated across all 16 B-columns)
    const int half = lane >> 4;     // which K-half this lane supplies
    const int m    = lane & 15;     // A-row within the 16-row tile
    const _Float16* qbase = f1h + ((size_t)pix << 8);   // feat1 pixel vector
    v8f acc0 = {};                  // scores for s = 0..15   (rows r+8*half)
    v8f acc1 = {};                  // scores for s = 16..31
    #pragma unroll
    for (int kb = 0; kb < CC; kb += 32) {
        v8h l0 = *(const v8h*)&smem[m * SPAD + kb + half * 8];
        v8h h0 = *(const v8h*)&smem[m * SPAD + kb + 16 + half * 8];
        v8h l1 = *(const v8h*)&smem[(m + 16) * SPAD + kb + half * 8];
        v8h h1 = *(const v8h*)&smem[(m + 16) * SPAD + kb + 16 + half * 8];
        const v8h* qp = (const v8h*)(qbase + kb + half * 8);
        v8h ql = qp[0], qh = qp[2];
        v16h a0, a1, bq;
        #pragma unroll
        for (int j = 0; j < 8; ++j) {
            a0[j] = l0[j]; a0[j + 8] = h0[j];
            a1[j] = l1[j]; a1[j + 8] = h1[j];
            bq[j] = ql[j]; bq[j + 8] = qh[j];
        }
        acc0 = __builtin_amdgcn_wmma_f32_16x16x32_f16(false, a0, false, bq,
                                                      (short)0, acc0, false, false);
        acc1 = __builtin_amdgcn_wmma_f32_16x16x32_f16(false, a1, false, bq,
                                                      (short)0, acc1, false, false);
    }

    // ---- softmax over 32 scores (lane holds s in {8*half..}, {16+8*half..})
    float mx = -3.0e38f;
    #pragma unroll
    for (int r = 0; r < 8; ++r) mx = fmaxf(mx, fmaxf(acc0[r], acc1[r]));
    mx = fmaxf(mx, __shfl_xor(mx, 16, 32));
    float e0[8], e1[8], ssum = 0.0f;
    #pragma unroll
    for (int r = 0; r < 8; ++r) {
        e0[r] = __expf(acc0[r] - mx);
        e1[r] = __expf(acc1[r] - mx);
        ssum += e0[r] + e1[r];
    }
    ssum += __shfl_xor(ssum, 16, 32);
    const float isum = 1.0f / ssum;
    if (m == 0) {
        #pragma unroll
        for (int r = 0; r < 8; ++r) {
            wgt[half * 8 + r]      = e0[r] * isum;
            wgt[16 + half * 8 + r] = e1[r] * isum;
        }
    }
    __syncthreads();

    // ---- phase 2: out[c] = sum_s w[s]*sampled[s][c]; lane owns 8 channels
    float accc[8] = {0.f, 0.f, 0.f, 0.f, 0.f, 0.f, 0.f, 0.f};
    #pragma unroll 4
    for (int s = 0; s < SS; ++s) {
        const float ws = wgt[s];
        v8h rv = *(const v8h*)&smem[s * SPAD + lane * 8];
        #pragma unroll
        for (int j = 0; j < 8; ++j) accc[j] += ws * (float)rv[j];
    }
    float* op = out + (((size_t)n * CC + lane * 8) * (HH * WW) + (h << 6) + w);
    #pragma unroll
    for (int j = 0; j < 8; ++j) op[j * (HH * WW)] = accc[j];
}

// ---------------------------------------------------------------------------
extern "C" void kernel_launch(void* const* d_in, const int* in_sizes, int n_in,
                              void* d_out, int out_size, void* d_ws, size_t ws_size,
                              hipStream_t stream) {
    const float* feat1 = (const float*)d_in[0];
    const float* feat2 = (const float*)d_in[1];
    const float* P1    = (const float*)d_in[2];
    const float* P2    = (const float*)d_in[3];
    float* out = (float*)d_out;

    const int Nb    = in_sizes[2] / 12;            // batches (N=2)
    const int total = Nb * HH * WW * CC;           // elements per feature tensor

    char* ws = (char*)d_ws;
    float*    Fm  = (float*)ws;                              // Nb*9 floats
    _Float16* f1h = (_Float16*)(ws + 256);                   // NHWC f16
    _Float16* f2h = (_Float16*)(ws + 256 + (size_t)total * sizeof(_Float16));

    epi_compute_F<<<1, 32, 0, stream>>>(P1, P2, Fm, Nb);
    epi_nchw_to_nhwc_f16<<<(total + 255) / 256, 256, 0, stream>>>(feat1, f1h, total);
    epi_nchw_to_nhwc_f16<<<(total + 255) / 256, 256, 0, stream>>>(feat2, f2h, total);
    epi_attn<<<Nb * HH * WW, 32, 0, stream>>>(f1h, f2h, Fm, out);
}
